// MHSA_3624952398502
// MI455X (gfx1250) — matrix operand order repaired and checked by reading, not compiled, not run on hardware
//
#include <hip/hip_runtime.h>
#include <hip/hip_bf16.h>

// ---------------------------------------------------------------------------
// MI455X (gfx1250) fused differential graph-attention block.
// Kernel 1: one-time f32->f16 conversion of all weight tensors into d_ws
//           (Wq, Wk, Wv, rpe, Wp) -- removes per-fragment cvt/perm VALU from
//           the hot kernel.
// Kernel 2: one workgroup per (batch, time, head); 64 threads = 2 wave32
//           waves (one per differential-attention branch). All matmuls on
//           v_wmma_f32_16x16x32_f16. V=25 padded to 32 -> 2x2 tiles; HD=32 =
//           1 K-step; C_IN=128 = 4 K-steps.
// ---------------------------------------------------------------------------

typedef _Float16 f16;
typedef __attribute__((ext_vector_type(16))) _Float16 v16h;
typedef __attribute__((ext_vector_type(8)))  float    v8f;

union FragH { v16h h; unsigned int u[8]; };

__device__ inline unsigned int pk2(float a, float b) {
  union { _Float16 h[2]; unsigned int u; } t;
  t.h[0] = (_Float16)a; t.h[1] = (_Float16)b;
  return t.u;
}

__device__ inline v8f wmma_f16(v16h a, v16h b, v8f c) {
  // D = A(16x32 f16) * B(32x16 f16) + C(16x16 f32)
  return __builtin_amdgcn_wmma_f32_16x16x32_f16(false, a, false, b,
                                                (short)0, c, false, false);
}

__device__ inline v8f zero8() {
  v8f z = {0.f,0.f,0.f,0.f,0.f,0.f,0.f,0.f};
  return z;
}

__device__ inline void wavefence() {
  // keep same-wave cross-lane LDS traffic ordered (HW keeps DS in order per
  // wave; this stops the compiler from reordering around it)
  __builtin_amdgcn_wave_barrier();
  __asm__ volatile("" ::: "memory");
}

// A fragment (16x32 f16) from row-major f16 LDS buffer.
// Layout used here: lane L -> row = L%16; lane half picks K 0-15 / 16-31 (contiguous).
__device__ inline v16h loadA_f16(const f16* buf, int ld, int rowbase, int kbase) {
  const int lane = threadIdx.x & 31;
  const int half = lane >> 4;
  const unsigned int* p =
      (const unsigned int*)(buf + (rowbase + (lane & 15)) * ld + kbase);
  FragH f;
#pragma unroll
  for (int j = 0; j < 8; ++j) f.u[j] = p[half * 8 + j];
  return f.h;
}

// A fragment from row-major f32 LDS buffer (K = 32, converted to f16).
__device__ inline v16h loadA_f32lds(const float* buf, int ld, int rowbase) {
  const int lane = threadIdx.x & 31;
  const int half = lane >> 4;
  const float* p = buf + (rowbase + (lane & 15)) * ld;
  FragH f;
#pragma unroll
  for (int j = 0; j < 8; ++j) {
    int c = (half * 8 + j) * 2;
    f.u[j] = pk2(p[c], p[c + 1]);
  }
  return f.h;
}

// B fragment (32x16 f16): B[k][col] = buf[col][k], buf row-major [col][k] f16
// in LDS. lane L -> col = L%16; half picks K 0-15 / 16-31 (contiguous).
__device__ inline v16h loadB_f16(const f16* buf, int ld, int colbase) {
  const int lane = threadIdx.x & 31;
  const int half = lane >> 4;
  const unsigned int* p =
      (const unsigned int*)(buf + (colbase + (lane & 15)) * ld);
  FragH f;
#pragma unroll
  for (int j = 0; j < 8; ++j) f.u[j] = p[half * 8 + j];
  return f.h;
}

// B fragment from GLOBAL f16 (pre-converted weights in d_ws), row-major
// [col][k]: B[k][col] = base[col*ld + kbase + k]. Contiguous 32B per lane.
__device__ inline v16h loadB_f16g(const f16* __restrict__ base, int ld,
                                  int colbase, int kbase, int colmax) {
  const int lane = threadIdx.x & 31;
  int col = colbase + (lane & 15);
  if (col >= colmax) col = colmax - 1;  // clamp (rpe has only 25 rows)
  const int half = lane >> 4;
  const unsigned int* p =
      (const unsigned int*)(base + (size_t)col * ld + kbase);
  FragH f;
#pragma unroll
  for (int j = 0; j < 8; ++j) f.u[j] = p[half * 8 + j];
  return f.h;
}

// C/D layout: col = lane%16, row = vgpr j + 8*(lane/16).
__device__ inline void storeD_f32(float* buf, int ld, int rowbase, int colbase,
                                  v8f acc) {
  const int lane = threadIdx.x & 31;
  const int col  = colbase + (lane & 15);
  const int half = lane >> 4;
#pragma unroll
  for (int j = 0; j < 8; ++j)
    buf[(rowbase + j + 8 * half) * ld + col] = acc[j];
}

__device__ inline void storeD_f16(f16* buf, int ld, int rowbase, int colbase,
                                  v8f acc) {
  const int lane = threadIdx.x & 31;
  const int col  = colbase + (lane & 15);
  const int half = lane >> 4;
#pragma unroll
  for (int j = 0; j < 8; ++j)
    buf[(rowbase + j + 8 * half) * ld + col] = (f16)acc[j];
}

__device__ inline void storeD_f16T(f16* buf, int ld, int rowbase, int colbase,
                                   v8f acc) {  // transposed: buf[col][row]
  const int lane = threadIdx.x & 31;
  const int col  = colbase + (lane & 15);
  const int half = lane >> 4;
#pragma unroll
  for (int j = 0; j < 8; ++j)
    buf[col * ld + rowbase + j + 8 * half] = (f16)acc[j];
}

// One 32x32 projection: out[m][c] (or transposed) = sum_k xT[m][k]*W16[wrow+c][k]
// W16: pre-converted f16 weights, row-major (rows of 128).
__device__ inline void project_mat(const f16* xT, const f16* __restrict__ W16,
                                   int wrow, f16* outbuf, bool transposed) {
#pragma unroll
  for (int ti = 0; ti < 2; ++ti) {
#pragma unroll
    for (int tc = 0; tc < 2; ++tc) {
      v8f acc = zero8();
#pragma unroll
      for (int ks = 0; ks < 4; ++ks) {
        v16h a = loadA_f16(xT, 128, ti * 16, ks * 32);
        v16h b = loadB_f16g(W16 + (size_t)wrow * 128, 128, tc * 16, ks * 32, 32);
        acc = wmma_f16(a, b, acc);
      }
      if (transposed) storeD_f16T(outbuf, 32, ti * 16, tc * 16, acc);
      else            storeD_f16 (outbuf, 32, ti * 16, tc * 16, acc);
    }
  }
}

// ---------------------------------------------------------------------------
// Prep: f32 -> f16 weight conversion into d_ws (runs once per launch; tiny).
// ---------------------------------------------------------------------------
__global__ __launch_bounds__(256)
void cvt_f32_to_f16_kernel(const float* __restrict__ src, f16* __restrict__ dst,
                           int n) {
  int i = blockIdx.x * 256 + threadIdx.x;
  if (i < n) dst[i] = (f16)src[i];
}

// ws layout (f16 element offsets)
#define WS_WQ16   0          // 384*128 = 49152
#define WS_WK16   49152      // 384*128 = 49152
#define WS_WV16   98304      // 192*128 = 24576
#define WS_RPE16  122880     // 25*192  = 4800
#define WS_WP16   127680     // 6*32*32 = 6144
#define WS_TOTAL  133824

__global__ __launch_bounds__(64)
void dattn_fused_kernel(const float* __restrict__ x, const float* __restrict__ e,
                        const f16* __restrict__ w16,   // prepped weights in ws
                        const float* __restrict__ w1, const float* __restrict__ outer,
                        const float* __restrict__ alpha_p,
                        const float* __restrict__ lq1, const float* __restrict__ lk1,
                        const float* __restrict__ lq2, const float* __restrict__ lk2,
                        const float* __restrict__ bp,
                        const int* __restrict__ hops, float* __restrict__ out) {
  constexpr int T = 128, V = 25, DIM = 192, HD = 32, CIN = 128;
  constexpr float SCALE = 0.17677669529663687f;  // 32^-0.5

  const int t = blockIdx.x, b = blockIdx.y, h = blockIdx.z;
  const int tid  = threadIdx.x;
  const int wid  = tid >> 5;   // 0: q1 branch, 1: q2 branch
  const int lane = tid & 31;

  const f16* Wq16  = w16 + WS_WQ16;
  const f16* Wk16  = w16 + WS_WK16;
  const f16* Wv16  = w16 + WS_WV16;
  const f16* rpe16 = w16 + WS_RPE16;
  const f16* Wp16  = w16 + WS_WP16;

  __shared__ __align__(16) f16 xT[32][128];         // x slice, transposed, f16
  __shared__ __align__(16) f16 q1b[32][32], q2b[32][32];
  __shared__ __align__(16) f16 k1b[32][32], k2b[32][32];
  __shared__ __align__(16) f16 ekb[32][32];         // e_k slice [m][c]
  __shared__ __align__(16) f16 vbT[32][32];         // v transposed [c][m]
  __shared__ __align__(16) float Sb[2][32][32];     // scores / xs scratch
  __shared__ __align__(16) float Pb[2][32][32];     // rpe-term P
  __shared__ __align__(16) f16 Ab[2][32][32];       // alpha*attn + outer (f16)
  __shared__ __align__(16) float x2b[32][32];       // branch-2 result
  __shared__ __align__(16) float ob[32][32];        // output staging [d][n]
  __shared__ int hopsb[V * V];

  // ---- stage inputs -------------------------------------------------------
  const size_t xbase = (size_t)b * CIN * (T * V) + (size_t)t * V;
  for (int idx = tid; idx < 32 * CIN; idx += 64) {
    int m = idx & 31, c = idx >> 5;
    xT[m][c] = (m < V) ? (f16)x[xbase + (size_t)c * (T * V) + m] : (f16)0.f;
  }
  for (int idx = tid; idx < V * V; idx += 64) hopsb[idx] = hops[idx];
  if (wid == 0) {
    const size_t ebase = ((size_t)b * DIM + h * HD) * (T * V) + (size_t)t * V;
    for (int idx = lane; idx < 32 * 32; idx += 32) {
      int m = idx & 31, c = idx >> 5;
      ekb[m][c] = (m < V) ? (f16)e[ebase + (size_t)c * (T * V) + m] : (f16)0.f;
    }
  }
  __syncthreads();

  // ---- projections (WMMA GEMM, K=128, f16 weights from ws) ----------------
  if (wid == 0) {
    project_mat(&xT[0][0], Wq16, h * HD,        &q1b[0][0], false);
    project_mat(&xT[0][0], Wk16, h * HD,        &k1b[0][0], false);
  } else {
    project_mat(&xT[0][0], Wq16, DIM + h * HD,  &q2b[0][0], false);
    project_mat(&xT[0][0], Wk16, DIM + h * HD,  &k2b[0][0], false);
    project_mat(&xT[0][0], Wv16, h * HD,        &vbT[0][0], true);
  }
  __syncthreads();

  const float alpha = alpha_p[0];
  const f16* qb = wid ? &q2b[0][0] : &q1b[0][0];
  const f16* kb = wid ? &k2b[0][0] : &k1b[0][0];
  float (*Sw)[32] = Sb[wid];
  float (*Pw)[32] = Pb[wid];
  f16   (*Aw)[32] = Ab[wid];

  // d[m] = sum_c w1[h,c] * e_k[m,c]  (column-broadcast term, used as C init)
  float dcol0 = 0.f, dcol1 = 0.f;
  {
    const int m0 = lane & 15;
#pragma unroll 8
    for (int c = 0; c < HD; ++c) {
      float wc = w1[h * HD + c];
      dcol0 += wc * (float)ekb[m0][c];
      dcol1 += wc * (float)ekb[m0 + 16][c];
    }
  }

  // ---- scores: acc = q*e_k^T + q*k^T + d ; P = q*rpe_h^T ------------------
#pragma unroll
  for (int ti = 0; ti < 2; ++ti) {
    v16h aq = loadA_f16(qb, 32, ti * 16, 0);
#pragma unroll
    for (int tj = 0; tj < 2; ++tj) {
      const float dv = tj ? dcol1 : dcol0;
      v8f c0;
#pragma unroll
      for (int j = 0; j < 8; ++j) c0[j] = dv;
      v8f s = wmma_f16(aq, loadB_f16(&ekb[0][0], 32, tj * 16), c0);
      s = wmma_f16(aq, loadB_f16(kb, 32, tj * 16), s);
      storeD_f32(&Sw[0][0], 32, ti * 16, tj * 16, s);
      v8f p = wmma_f16(aq, loadB_f16g(rpe16, DIM, tj * 16, h * HD, V), zero8());
      storeD_f32(&Pw[0][0], 32, ti * 16, tj * 16, p);
    }
  }
  __syncthreads();

  // ---- softmax rows + hops-gather of P + (alpha*attn + outer) -> f16 ------
  {
    const int r = lane;
    if (r < V) {
      const int* hr = hopsb + r * V;
      float mx = -1e30f;
      for (int m = 0; m < V; ++m) {
        float s = (Sw[r][m] + Pw[r][hr[m]]) * SCALE;
        Sw[r][m] = s;
        mx = fmaxf(mx, s);
      }
      float sum = 0.f;
      for (int m = 0; m < V; ++m) {
        float ev = __expf(Sw[r][m] - mx);
        sum += ev;
        Sw[r][m] = ev;
      }
      const float inv = alpha / sum;
      for (int m = 0; m < V; ++m)
        Aw[r][m] = (f16)(Sw[r][m] * inv + outer[(h * V + r) * V + m]);
      for (int m = V; m < 32; ++m) Aw[r][m] = (f16)0.f;   // K padding
    } else {
      for (int m = 0; m < 32; ++m) Aw[r][m] = (f16)0.f;   // padded rows
    }
  }
  __syncthreads();

  // ---- apply: X = (alpha*attn + outer) @ v  (K = joints, padded to 32) ----
  v8f X[2][2];
#pragma unroll
  for (int ti = 0; ti < 2; ++ti) {
    v16h a = loadA_f16(&Aw[0][0], 32, ti * 16, 0);
#pragma unroll
    for (int tc = 0; tc < 2; ++tc)
      X[ti][tc] = wmma_f16(a, loadB_f16(&vbT[0][0], 32, tc * 16), zero8());
  }
  if (wid == 1) {
#pragma unroll
    for (int ti = 0; ti < 2; ++ti)
#pragma unroll
      for (int tc = 0; tc < 2; ++tc)
        storeD_f32(&x2b[0][0], 32, ti * 16, tc * 16, X[ti][tc]);
  }
  __syncthreads();

  // ---- wave0: xs = X1 - lam*X2 ; out = xs @ Wp_h^T + bp -------------------
  if (wid == 0) {
    float l1 = 0.f, l2 = 0.f;
#pragma unroll
    for (int i = 0; i < 16; ++i) {
      l1 += lq1[i] * lk1[i];
      l2 += lq2[i] * lk2[i];
    }
    const float LAMBDA_INIT = 0.8f - 0.6f * __expf(-1.8f);
    const float lam = __expf(l1) - __expf(l2) + LAMBDA_INIT;

    float (*xsb)[32] = Sb[0];  // reuse scores scratch
    const int col = lane & 15, half = lane >> 4;
#pragma unroll
    for (int ti = 0; ti < 2; ++ti)
#pragma unroll
      for (int tc = 0; tc < 2; ++tc)
#pragma unroll
        for (int j = 0; j < 8; ++j) {
          int rr = ti * 16 + j + 8 * half;
          int cc = tc * 16 + col;
          xsb[rr][cc] = X[ti][tc][j] - lam * x2b[rr][cc];
        }
    wavefence();

#pragma unroll
    for (int ti = 0; ti < 2; ++ti) {
      v16h a = loadA_f32lds(&xsb[0][0], 32, ti * 16);
#pragma unroll
      for (int td = 0; td < 2; ++td) {
        v8f o = wmma_f16(a, loadB_f16g(Wp16 + (size_t)h * HD * HD, HD,
                                       td * 16, 0, HD), zero8());
        const int dl = td * 16 + col;
        const float bpv = bp[h * HD + dl];
#pragma unroll
        for (int j = 0; j < 8; ++j)
          ob[dl][ti * 16 + j + 8 * half] = o[j] + bpv;
      }
    }
    wavefence();

    const size_t obase = (((size_t)b * DIM + h * HD) * T + t) * V;
    if (lane < V) {
#pragma unroll 4
      for (int d = 0; d < HD; ++d)
        out[obase + (size_t)d * (T * V) + lane] = ob[d][lane];
    }
  }
}

extern "C" void kernel_launch(void* const* d_in, const int* in_sizes, int n_in,
                              void* d_out, int out_size, void* d_ws, size_t ws_size,
                              hipStream_t stream) {
  (void)in_sizes; (void)n_in; (void)ws_size; (void)out_size;
  const float* x     = (const float*)d_in[0];
  const float* e     = (const float*)d_in[1];
  const float* Wv    = (const float*)d_in[2];
  const float* Wk    = (const float*)d_in[3];
  const float* Wq    = (const float*)d_in[4];
  const float* rpe   = (const float*)d_in[5];
  const float* w1    = (const float*)d_in[6];
  const float* outer = (const float*)d_in[7];
  const float* alpha = (const float*)d_in[8];
  const float* lq1   = (const float*)d_in[9];
  const float* lk1   = (const float*)d_in[10];
  const float* lq2   = (const float*)d_in[11];
  const float* lk2   = (const float*)d_in[12];
  const float* Wp    = (const float*)d_in[13];
  const float* bp    = (const float*)d_in[14];
  const int*   hops  = (const int*)d_in[15];
  float* out = (float*)d_out;
  f16*   w16 = (f16*)d_ws;

  // ---- prep: convert all weights to f16 once per launch -------------------
  {
    const int nq = 384 * 128, nk = 384 * 128, nv = 192 * 128;
    const int nr = 25 * 192, np = 6 * 32 * 32;
    hipLaunchKernelGGL(cvt_f32_to_f16_kernel, dim3((nq + 255) / 256), dim3(256),
                       0, stream, Wq, w16 + WS_WQ16, nq);
    hipLaunchKernelGGL(cvt_f32_to_f16_kernel, dim3((nk + 255) / 256), dim3(256),
                       0, stream, Wk, w16 + WS_WK16, nk);
    hipLaunchKernelGGL(cvt_f32_to_f16_kernel, dim3((nv + 255) / 256), dim3(256),
                       0, stream, Wv, w16 + WS_WV16, nv);
    hipLaunchKernelGGL(cvt_f32_to_f16_kernel, dim3((nr + 255) / 256), dim3(256),
                       0, stream, rpe, w16 + WS_RPE16, nr);
    hipLaunchKernelGGL(cvt_f32_to_f16_kernel, dim3((np + 255) / 256), dim3(256),
                       0, stream, Wp, w16 + WS_WP16, np);
  }

  // ---- fused attention ----------------------------------------------------
  dim3 grid(128 /*T*/, 64 /*N*/, 6 /*H*/);
  dim3 block(64);  // 2 wave32 waves: one per differential-attention branch
  hipLaunchKernelGGL(dattn_fused_kernel, grid, block, 0, stream,
                     x, e, w16, w1, outer, alpha,
                     lq1, lk1, lq2, lk2, bp, hops, out);
}
